// CausalSelfAttention_472446403149
// MI455X (gfx1250) — matrix operand order repaired and checked by reading, not compiled, not run on hardware
//
#include <hip/hip_runtime.h>

// ---------------------------------------------------------------------------
// CDNA5 (gfx1250, wave32) causal self-attention forward.
// Pipeline: QKV GEMM (bf16 WMMA) -> flash attention (bf16 WMMA + TDM async
//           tile loads, f32 softmax) -> proj GEMM (bf16 WMMA + TDM A tiles).
// ---------------------------------------------------------------------------

typedef __bf16 bf16_t;
typedef __attribute__((ext_vector_type(16))) __bf16 v16bf;
typedef __attribute__((ext_vector_type(8)))  __bf16 v8bf;
typedef __attribute__((ext_vector_type(8)))  float  v8f;

typedef unsigned int u32;
typedef __attribute__((ext_vector_type(4))) u32 u32x4;
typedef __attribute__((ext_vector_type(4))) int i32x4;
typedef __attribute__((ext_vector_type(8))) int i32x8;

#define T_SEQ   2048
#define N_HEADS 16
#define HEAD_D  64
#define EMBED   1024
#define M_ROWS  8192      // B*T
#define N_QKV   3072

union AFrag { v16bf v; v8bf h[2]; };

// A-matrix 16x32 bf16 fragment (wave32):
//  lane<16 : row=lane,    K in {0..7, 16..23}
//  lane>=16: row=lane-16, K in {8..15, 24..31}
__device__ __forceinline__ v16bf load_frag_a(const bf16_t* tile, int lane, int stride) {
  const int hl = lane >> 4, r = lane & 15;
  AFrag f;
  f.h[0] = *(const v8bf*)(tile + r * stride + hl * 8);
  f.h[1] = *(const v8bf*)(tile + r * stride + 16 + hl * 8);
  return f.v;
}

// B-matrix 32x16 bf16 fragment from an N-major (transposed) tile tileT[n][k]:
//  lane<16 : col=lane,    K in {0..7, 16..23}
//  lane>=16: col=lane-16, K in {8..15, 24..31}
__device__ __forceinline__ v16bf load_frag_b(const bf16_t* tileT, int lane, int stride) {
  const int hl = lane >> 4, c = lane & 15;
  AFrag f;
  f.h[0] = *(const v8bf*)(tileT + c * stride + hl * 8);
  f.h[1] = *(const v8bf*)(tileT + c * stride + 16 + hl * 8);
  return f.v;
}

__device__ __forceinline__ v8f wmma_bf16(v16bf a, v16bf b, v8f c) {
  return __builtin_amdgcn_wmma_f32_16x16x32_bf16(false, a, false, b, (short)0, c,
                                                 false, false);
}

__device__ __forceinline__ v8f vzero8() {
  v8f z = {0.f, 0.f, 0.f, 0.f, 0.f, 0.f, 0.f, 0.f};
  return z;
}

// ---------------------------------------------------------------------------
// Tensor Data Mover: 2D tile load, global -> LDS, bf16 elements (data_size=2),
// with LDS row padding.  Descriptor per CDNA5 ISA 8.3/8.4:
//   group0: count=1 | lds_addr | global_addr(57b) | type=2
//   group1: data_size=1(2B) | pad_enable | pad_interval | pad_amount |
//           tensor_dim0/1 | tile_dim0/1 | tensor_dim0_stride
// pad_int_code: LDS bytes between pads = 8 << code; pad_amt_code: pad bytes =
// 4*(code+1).  Toolchain here is the 6-arg clang-23 builtin:
//   (u32x4 g0, i32x8 g1, i32x4 g2, i32x4 g3, i32x8 extra, i32 cpol)
// ---------------------------------------------------------------------------
__device__ __forceinline__ void tdm_load_2d_bf16(
    u32 lds_off, const void* gaddr, u32 tensor_d0, u32 tensor_d1,
    u32 stride0, u32 tile_d0, u32 tile_d1, u32 pad_int_code, u32 pad_amt_code) {
  const unsigned long long ga = (unsigned long long)gaddr;
  u32x4 g0;
  g0[0] = 1u;                                        // count=1, user desc
  g0[1] = lds_off;                                   // LDS byte address
  g0[2] = (u32)(ga & 0xffffffffu);                   // global_addr[31:0]
  g0[3] = (u32)((ga >> 32) & 0x01ffffffu) | (2u << 30);  // [56:32] | type=2
  i32x8 g1;
  g1[0] = (int)((1u << 16)                           // data_size = 2 bytes
              | (1u << 20)                           // pad_enable
              | (pad_int_code << 22)
              | (pad_amt_code << 25));
  g1[1] = (int)((tensor_d0 & 0xffffu) << 16);        // tensor_dim0[15:0]
  g1[2] = (int)(((tensor_d0 >> 16) & 0xffffu) | ((tensor_d1 & 0xffffu) << 16));
  g1[3] = (int)(((tensor_d1 >> 16) & 0xffffu) | ((tile_d0 & 0xffffu) << 16));
  g1[4] = (int)(tile_d1 & 0xffffu);                  // tile_dim1, tile_dim2=0
  g1[5] = (int)stride0;                              // tensor_dim0_stride[31:0]
  g1[6] = 0;                                         // stride[47:32], dim1_stride
  g1[7] = 0;
  const i32x4 gz4 = {0, 0, 0, 0};
  const i32x8 gz8 = {0, 0, 0, 0, 0, 0, 0, 0};
  __builtin_amdgcn_tensor_load_to_lds(g0, g1, gz4, gz4, gz8, 0);
}

__device__ __forceinline__ u32 lds_offset_of(const void* p) {
  return (u32)(unsigned long long)p;   // low 32 bits of generic LDS ptr = offset
}

// ---------------------------------------------------------------------------
// Kernel 1: QKV GEMM.  C[m,n] = x[m,:] @ w_qkv[:,n] + b_qkv[n]
// q,k scattered as bf16 [B,H,T,D]; v stored TRANSPOSED as [B,H,D,T] so the
// attention kernel can TDM-copy V tiles without a transpose.
// Q gets the 1/sqrt(D) = 0.125 attention scale folded in.
// ---------------------------------------------------------------------------
#define GK_LDA 40   // padded LDS stride (elements), multiple of 8
#define GK_LDB 40

__global__ __launch_bounds__(256) void qkv_gemm_kernel(
    const float* __restrict__ x,      // [8192, 1024]
    const float* __restrict__ w,      // [1024, 3072]
    const float* __restrict__ bias,   // [3072]
    bf16_t* __restrict__ qo,          // [B,H,T,D]
    bf16_t* __restrict__ ko,          // [B,H,T,D]
    bf16_t* __restrict__ vo) {        // [B,H,D,T]  (transposed)
  __shared__ bf16_t As[128 * GK_LDA];
  __shared__ bf16_t Bs[128 * GK_LDB];   // N-major: Bs[n][k]

  const int tid  = threadIdx.x;
  const int lane = tid & 31;
  const int wave = tid >> 5;            // 0..7
  const int wm   = wave >> 1;
  const int wn   = wave & 1;
  const int m0   = blockIdx.x * 128;
  const int n0   = blockIdx.y * 128;

  v8f acc[2][4];
#pragma unroll
  for (int i = 0; i < 2; ++i)
#pragma unroll
    for (int j = 0; j < 4; ++j) acc[i][j] = vzero8();

  for (int kk = 0; kk < EMBED; kk += 32) {
    __syncthreads();
    {
      const int r = tid >> 3;
      const int c = (tid & 7) * 4;
#pragma unroll
      for (int p = 0; p < 4; ++p) {
        const int row = r + p * 32;
        const float4 d = *(const float4*)(x + (size_t)(m0 + row) * EMBED + kk + c);
        bf16_t* dst = As + row * GK_LDA + c;
        dst[0] = (bf16_t)d.x; dst[1] = (bf16_t)d.y;
        dst[2] = (bf16_t)d.z; dst[3] = (bf16_t)d.w;
      }
    }
    {
      const int kq = tid >> 5;
      const int nq = (tid & 31) * 4;
#pragma unroll
      for (int p = 0; p < 4; ++p) {
        const int krow = kq + p * 8;
        const float4 d = *(const float4*)(w + (size_t)(kk + krow) * N_QKV + n0 + nq);
        Bs[(nq + 0) * GK_LDB + krow] = (bf16_t)d.x;
        Bs[(nq + 1) * GK_LDB + krow] = (bf16_t)d.y;
        Bs[(nq + 2) * GK_LDB + krow] = (bf16_t)d.z;
        Bs[(nq + 3) * GK_LDB + krow] = (bf16_t)d.w;
      }
    }
    __syncthreads();

    v16bf a[2], b[4];
    a[0] = load_frag_a(As + (wm * 32 + 0)  * GK_LDA, lane, GK_LDA);
    a[1] = load_frag_a(As + (wm * 32 + 16) * GK_LDA, lane, GK_LDA);
#pragma unroll
    for (int j = 0; j < 4; ++j)
      b[j] = load_frag_b(Bs + (wn * 64 + j * 16) * GK_LDB, lane, GK_LDB);
#pragma unroll
    for (int i = 0; i < 2; ++i)
#pragma unroll
      for (int j = 0; j < 4; ++j)
        acc[i][j] = wmma_bf16(a[i], b[j], acc[i][j]);
  }

  // ---- epilogue: bias, Q-scale, scatter ----
  const int hl = lane >> 4, cx = lane & 15;
#pragma unroll
  for (int j = 0; j < 4; ++j) {
    const int n     = n0 + wn * 64 + j * 16 + cx;
    const int which = n >> 10;           // 0=q 1=k 2=v
    const int c     = n & 1023;
    const int h     = c >> 6, d = c & 63;
    const float bsc = bias[n];
#pragma unroll
    for (int i = 0; i < 2; ++i) {
      const int mb = m0 + wm * 32 + i * 16 + hl * 8;   // 8 consecutive rows
      const int bb = mb >> 11;
      const int t0 = mb & (T_SEQ - 1);
      if (which == 2) {
        // V^T [B,H,D,T]: consecutive rows -> consecutive t: one packed store.
        v8bf pk;
#pragma unroll
        for (int r = 0; r < 8; ++r) pk[r] = (bf16_t)(acc[i][j][r] + bsc);
        *(v8bf*)(vo + (((size_t)bb * N_HEADS + h) * HEAD_D + d) * T_SEQ + t0) = pk;
      } else {
        bf16_t* dst = (which == 0) ? qo : ko;
        const float mul = (which == 0) ? 0.125f : 1.0f;
#pragma unroll
        for (int r = 0; r < 8; ++r) {
          const float vv = (acc[i][j][r] + bsc) * mul;
          dst[(((size_t)bb * N_HEADS + h) * T_SEQ + t0 + r) * HEAD_D + d] = (bf16_t)vv;
        }
      }
    }
  }
}

// ---------------------------------------------------------------------------
// Kernel 2: flash attention (causal, online softmax), TDM-fed K/V tiles.
// Grid: (B*H, T/64). Workgroup = 128 thr = 4 waves; wave owns 16 query rows.
// Double-buffered 64-key blocks: wave0 issues tensor_load_to_lds for block
// kb+1 while all waves compute block kb.
// ---------------------------------------------------------------------------
#define AT_LDS 72   // padded stride (elements) = 64 + 8, produced by TDM pad

__global__ __launch_bounds__(128) void attn_kernel(
    const bf16_t* __restrict__ qg,   // [B,H,T,D], pre-scaled
    const bf16_t* __restrict__ kg,   // [B,H,T,D]
    const bf16_t* __restrict__ vtg,  // [B,H,D,T]  (transposed)
    bf16_t* __restrict__ og) {       // [B,T,H,D]
  __shared__ bf16_t Ks[2][64 * AT_LDS];     // K block, row-major [key][d]
  __shared__ bf16_t Vt[2][64 * AT_LDS];     // V block, [d][key]
  __shared__ bf16_t Ps[4][16 * AT_LDS];     // per-wave P scratch [row][key]

  const int tid  = threadIdx.x;
  const int lane = tid & 31;
  const int wave = tid >> 5;                // 0..3
  const int bh   = blockIdx.x;
  const int b    = bh >> 4, h = bh & 15;
  const int q0   = blockIdx.y * 64;
  const int qw   = q0 + wave * 16;

  const bf16_t* qbase  = qg  + (size_t)bh * T_SEQ * HEAD_D;
  const bf16_t* kbase  = kg  + (size_t)bh * T_SEQ * HEAD_D;
  const bf16_t* vtbase = vtg + (size_t)bh * HEAD_D * T_SEQ;

  v16bf qa0 = load_frag_a(qbase + (size_t)qw * HEAD_D, lane, HEAD_D);
  v16bf qa1 = load_frag_a(qbase + (size_t)qw * HEAD_D + 32, lane, HEAD_D);

  v8f o[4];
#pragma unroll
  for (int j = 0; j < 4; ++j) o[j] = vzero8();
  float m_i[8], l_i[8];
#pragma unroll
  for (int r = 0; r < 8; ++r) { m_i[r] = -1e30f; l_i[r] = 0.f; }

  const int hl = lane >> 4, cx = lane & 15;
  const int nkb = blockIdx.y + 1;           // causal: key blocks up to diagonal

  // Prime buffer 0 with key block 0.  K tile: 64 keys x 64 d from [T,D];
  // Vt tile: 64 d x 64 t from [D,T].  pad: every 128B (code 4) add 16B (code 3).
  if (wave == 0) {
    tdm_load_2d_bf16(lds_offset_of(&Ks[0][0]), kbase,
                     HEAD_D, T_SEQ, HEAD_D, 64, 64, 4, 3);
    tdm_load_2d_bf16(lds_offset_of(&Vt[0][0]), vtbase,
                     T_SEQ, HEAD_D, T_SEQ, 64, 64, 4, 3);
  }

  for (int kb = 0; kb < nkb; ++kb) {
    const int cur = kb & 1;
    const int kstart = kb * 64;
    __builtin_amdgcn_s_wait_tensorcnt(0);   // all waves: no-op unless issuer
    __syncthreads();                        // buffer `cur` ready for everyone
    if (wave == 0 && (kb + 1) < nkb) {
      const int ks2 = (kb + 1) * 64;
      tdm_load_2d_bf16(lds_offset_of(&Ks[1 - cur][0]),
                       kbase + (size_t)ks2 * HEAD_D,
                       HEAD_D, T_SEQ, HEAD_D, 64, 64, 4, 3);
      tdm_load_2d_bf16(lds_offset_of(&Vt[1 - cur][0]),
                       vtbase + ks2,
                       T_SEQ, HEAD_D, T_SEQ, 64, 64, 4, 3);
    }

    // ---- S = Q K^T : 4 key-tiles x 2 K-dim steps ----
    v8f s[4];
#pragma unroll
    for (int j = 0; j < 4; ++j) {
      s[j] = vzero8();
      const v16bf bk0 = load_frag_b(&Ks[cur][(j * 16) * AT_LDS], lane, AT_LDS);
      const v16bf bk1 = load_frag_b(&Ks[cur][(j * 16) * AT_LDS + 32], lane, AT_LDS);
      s[j] = wmma_bf16(qa0, bk0, s[j]);
      s[j] = wmma_bf16(qa1, bk1, s[j]);
    }

    // ---- causal mask + block row-max ----
    float mblk[8];
#pragma unroll
    for (int r = 0; r < 8; ++r) mblk[r] = -1e30f;
#pragma unroll
    for (int j = 0; j < 4; ++j) {
      const int key = kstart + j * 16 + cx;
#pragma unroll
      for (int r = 0; r < 8; ++r) {
        const int qrow = qw + hl * 8 + r;
        float sv = s[j][r];
        sv = (key <= qrow) ? sv : -1e30f;
        s[j][r] = sv;
        mblk[r] = fmaxf(mblk[r], sv);
      }
    }
#pragma unroll
    for (int off = 1; off < 16; off <<= 1)
#pragma unroll
      for (int r = 0; r < 8; ++r)
        mblk[r] = fmaxf(mblk[r], __shfl_xor(mblk[r], off, 32));

    // ---- online rescale ----
    float alpha[8];
#pragma unroll
    for (int r = 0; r < 8; ++r) {
      const float mnew = fmaxf(m_i[r], mblk[r]);
      alpha[r] = __expf(m_i[r] - mnew);
      m_i[r]   = mnew;
      l_i[r]  *= alpha[r];
    }
#pragma unroll
    for (int j = 0; j < 4; ++j)
#pragma unroll
      for (int r = 0; r < 8; ++r) o[j][r] *= alpha[r];

    // ---- P = exp(S - m), row sums, spill to per-wave LDS ----
    float rsum[8];
#pragma unroll
    for (int r = 0; r < 8; ++r) rsum[r] = 0.f;
#pragma unroll
    for (int j = 0; j < 4; ++j) {
#pragma unroll
      for (int r = 0; r < 8; ++r) {
        const float p = __expf(s[j][r] - m_i[r]);
        rsum[r] += p;
        Ps[wave][(hl * 8 + r) * AT_LDS + j * 16 + cx] = (bf16_t)p;
      }
    }
#pragma unroll
    for (int off = 1; off < 16; off <<= 1)
#pragma unroll
      for (int r = 0; r < 8; ++r) rsum[r] += __shfl_xor(rsum[r], off, 32);
#pragma unroll
    for (int r = 0; r < 8; ++r) l_i[r] += rsum[r];

    // ---- O += P V (per-wave LDS ops are in-order; no barrier needed) ----
    const v16bf pa0 = load_frag_a(&Ps[wave][0], lane, AT_LDS);
    const v16bf pa1 = load_frag_a(&Ps[wave][32], lane, AT_LDS);
#pragma unroll
    for (int j = 0; j < 4; ++j) {
      const v16bf bv0 = load_frag_b(&Vt[cur][(j * 16) * AT_LDS], lane, AT_LDS);
      const v16bf bv1 = load_frag_b(&Vt[cur][(j * 16) * AT_LDS + 32], lane, AT_LDS);
      o[j] = wmma_bf16(pa0, bv0, o[j]);
      o[j] = wmma_bf16(pa1, bv1, o[j]);
    }
  }

  // ---- finalize: O /= l, store [B,T,H,D] bf16 ----
#pragma unroll
  for (int j = 0; j < 4; ++j) {
#pragma unroll
    for (int r = 0; r < 8; ++r) {
      const int t = qw + hl * 8 + r;
      const int d = j * 16 + cx;
      const float val = o[j][r] / l_i[r];
      og[((size_t)b * T_SEQ + t) * EMBED + h * HEAD_D + d] = (bf16_t)val;
    }
  }
}

// ---------------------------------------------------------------------------
// Kernel 3: output projection.  out = ao @ w_proj + b_proj, f32 out.
// A tile (bf16) fetched by TDM, overlapped with B-tile f32->bf16 staging.
// ---------------------------------------------------------------------------
__global__ __launch_bounds__(256) void proj_gemm_kernel(
    const bf16_t* __restrict__ a,    // [8192, 1024] bf16
    const float* __restrict__ w,     // [1024, 1024]
    const float* __restrict__ bias,  // [1024]
    float* __restrict__ out) {       // [8192, 1024]
  __shared__ bf16_t As[128 * GK_LDA];
  __shared__ bf16_t Bs[128 * GK_LDB];

  const int tid  = threadIdx.x;
  const int lane = tid & 31;
  const int wave = tid >> 5;
  const int wm   = wave >> 1;
  const int wn   = wave & 1;
  const int m0   = blockIdx.x * 128;
  const int n0   = blockIdx.y * 128;

  v8f acc[2][4];
#pragma unroll
  for (int i = 0; i < 2; ++i)
#pragma unroll
    for (int j = 0; j < 4; ++j) acc[i][j] = vzero8();

  for (int kk = 0; kk < EMBED; kk += 32) {
    __syncthreads();   // everyone done reading As/Bs from previous step
    // A tile via TDM: 128 rows x 32 k bf16 from [8192,1024];
    // row = 64B (pad interval code 3), pad 16B (code 3) -> stride 40 elements.
    if (wave == 0) {
      tdm_load_2d_bf16(lds_offset_of(&As[0]),
                       a + (size_t)m0 * EMBED + kk,
                       EMBED, M_ROWS, EMBED, 32, 128, 3, 3);
    }
    // B tile transposed, f32 -> bf16 (overlaps the TDM copy)
    {
      const int kq = tid >> 5;
      const int nq = (tid & 31) * 4;
#pragma unroll
      for (int p = 0; p < 4; ++p) {
        const int krow = kq + p * 8;
        const float4 d = *(const float4*)(w + (size_t)(kk + krow) * EMBED + n0 + nq);
        Bs[(nq + 0) * GK_LDB + krow] = (bf16_t)d.x;
        Bs[(nq + 1) * GK_LDB + krow] = (bf16_t)d.y;
        Bs[(nq + 2) * GK_LDB + krow] = (bf16_t)d.z;
        Bs[(nq + 3) * GK_LDB + krow] = (bf16_t)d.w;
      }
    }
    __builtin_amdgcn_s_wait_tensorcnt(0);
    __syncthreads();

    v16bf af[2], bf[4];
    af[0] = load_frag_a(As + (wm * 32 + 0)  * GK_LDA, lane, GK_LDA);
    af[1] = load_frag_a(As + (wm * 32 + 16) * GK_LDA, lane, GK_LDA);
#pragma unroll
    for (int j = 0; j < 4; ++j)
      bf[j] = load_frag_b(Bs + (wn * 64 + j * 16) * GK_LDB, lane, GK_LDB);
#pragma unroll
    for (int i = 0; i < 2; ++i)
#pragma unroll
      for (int j = 0; j < 4; ++j)
        acc[i][j] = wmma_bf16(af[i], bf[j], acc[i][j]);
  }

  const int hl = lane >> 4, cx = lane & 15;
#pragma unroll
  for (int j = 0; j < 4; ++j) {
    const int n = n0 + wn * 64 + j * 16 + cx;
    const float bsc = bias[n];
#pragma unroll
    for (int i = 0; i < 2; ++i) {
#pragma unroll
      for (int r = 0; r < 8; ++r) {
        const int m = m0 + wm * 32 + i * 16 + hl * 8 + r;
        out[(size_t)m * EMBED + n] = acc[i][j][r] + bsc;
      }
    }
  }
}

// ---------------------------------------------------------------------------
extern "C" void kernel_launch(void* const* d_in, const int* in_sizes, int n_in,
                              void* d_out, int out_size, void* d_ws, size_t ws_size,
                              hipStream_t stream) {
  const float* x      = (const float*)d_in[0];
  const float* w_qkv  = (const float*)d_in[1];
  const float* b_qkv  = (const float*)d_in[2];
  const float* w_proj = (const float*)d_in[3];
  const float* b_proj = (const float*)d_in[4];
  float* out = (float*)d_out;

  // Workspace: q,k [B,H,T,D], v [B,H,D,T], attn-out [B,T,H,D] (all bf16).
  const size_t elems = (size_t)M_ROWS * EMBED;
  bf16_t* q  = (bf16_t*)d_ws;
  bf16_t* k  = q + elems;
  bf16_t* vt = k + elems;
  bf16_t* ao = vt + elems;

  qkv_gemm_kernel<<<dim3(M_ROWS / 128, N_QKV / 128), 256, 0, stream>>>(
      x, w_qkv, b_qkv, q, k, vt);
  attn_kernel<<<dim3(4 * N_HEADS, T_SEQ / 64), 128, 0, stream>>>(q, k, vt, ao);
  proj_gemm_kernel<<<dim3(M_ROWS / 128, EMBED / 128), 256, 0, stream>>>(
      ao, w_proj, b_proj, out);
}
